// SCAM_32134945309403
// MI455X (gfx1250) — matrix operand order repaired and checked by reading, not compiled, not run on hardware
//
#include <hip/hip_runtime.h>
#include <hip/hip_bf16.h>

typedef __attribute__((ext_vector_type(16))) __bf16 v16bf;
typedef __attribute__((ext_vector_type(8)))  float  v8f;
typedef __attribute__((ext_vector_type(4)))  unsigned int v4u;
typedef __attribute__((ext_vector_type(8)))  int  v8i_;
typedef __attribute__((ext_vector_type(4)))  int  v4i_;

#define B_   8
#define C_   512
#define N_   4096
#define C4_  2048

// LDS row strides (elements), all chosen so 16B-aligned chunk loads work
#define QS   528   // Qs/Ks [row][c] stride (bf16)
#define VS   80    // Vs [c][m] stride (bf16)    = 64 data + 8 DW pad
#define SS   68    // Ssc [row][key] stride (f32)
#define PSR  80    // Ps [row][key] stride (bf16)
#define OSS  528   // FFN O tile [t][c] stride (bf16) = 512 data + 8 DW pad
#define HSS  2064  // FFN hidden [t][d] stride (bf16)

union FragB16 { v16bf v; uint4 q[2]; };
union Pack8   { __bf16 h[8]; uint4 u; };

// ---------------------------------------------------------------- TDM helper
// 2D tile DMA: global (row-major, stride0 elems) -> LDS, optional LDS padding.
// flags = group1 dword0: data_size<<16 | pad_enable<<20 | pad_interval<<22 |
//                        pad_amount<<25   (workgroup_mask = 0: not in cluster)
__device__ inline void tdm_load_2d(unsigned lds_addr, const void* gptr,
                                   unsigned flags,
                                   unsigned tensor0, unsigned tensor1,
                                   unsigned tile0, unsigned tile1,
                                   unsigned stride0) {
  unsigned long long ga = (unsigned long long)(size_t)gptr;
  v4u g0 = { 1u,                                   // count=1, user mode
             lds_addr,                             // lds_addr[31:0]
             (unsigned)(ga & 0xFFFFFFFFu),         // global_addr lo
             (unsigned)((ga >> 32) & 0x01FFFFFFu) | (2u << 30) }; // hi | type=2
  v8i_ g1 = { (int)flags,
              (int)((tensor0 & 0xFFFFu) << 16),                        // [63:48] tdim0 lo
              (int)(((tensor0 >> 16) & 0xFFFFu) | ((tensor1 & 0xFFFFu) << 16)),
              (int)(((tensor1 >> 16) & 0xFFFFu) | ((tile0 & 0xFFFFu) << 16)),
              (int)(tile1 & 0xFFFFu),                                  // tile1 | tile2=0
              (int)stride0,                                            // dim0_stride lo32
              0,                                                       // hi16 | dim1_stride=0
              0 };
  v4i_ z4 = { 0, 0, 0, 0 };
#if __has_include(<hip/amd_detail/amd_gfx1250_TDM.h>)
  v8i_ z8 = { 0, 0, 0, 0, 0, 0, 0, 0 };
  __builtin_amdgcn_tensor_load_to_lds(g0, g1, z4, z4, z8, 0);
#else
  __builtin_amdgcn_tensor_load_to_lds(g0, g1, z4, z4, 0);
#endif
}

// A-matrix fragment (16x32 bf16): lane<16 holds row=l15, K = k0+{0..7,16..23};
// lane>=16 holds row=l15, K = k0+{8..15,24..31}.  Two b128 loads.
__device__ inline v16bf frag_a(const __bf16* base, int row0, int stride, int k0,
                               int l15, int hi) {
  FragB16 f;
  const __bf16* p = base + (size_t)(row0 + l15) * stride + k0 + hi * 8;
  f.q[0] = *reinterpret_cast<const uint4*>(p);
  f.q[1] = *reinterpret_cast<const uint4*>(p + 16);
  return f.v;
}

// B-matrix fragment (32x16 bf16): lane<16 holds col=l15, K = k0+{0..7,16..23};
// lane>=16 holds col=l15, K = k0+{8..15,24..31}.  Two b128 loads.
__device__ inline v16bf frag_b(const __bf16* base, int col0, int stride, int k0,
                               int l15, int hi) {
  FragB16 f;
  const __bf16* p = base + (size_t)(col0 + l15) * stride + k0 + hi * 8;
  f.q[0] = *reinterpret_cast<const uint4*>(p);
  f.q[1] = *reinterpret_cast<const uint4*>(p + 16);
  return f.v;
}

// ------------------------------------------------- f32 -> bf16 prep (w1,w2,k,v)
__global__ __launch_bounds__(256)
void cvt_kernel(const float* __restrict__ w1, const float* __restrict__ w2,
                const float* __restrict__ k,  const float* __restrict__ v,
                __bf16* __restrict__ w1b, __bf16* __restrict__ w2b,
                __bf16* __restrict__ kb,  __bf16* __restrict__ vb) {
  const size_t nw  = (size_t)C4_ * C_;
  const size_t nkv = (size_t)B_ * C_ * N_;
  size_t i = (size_t)blockIdx.x * blockDim.x + threadIdx.x;
  if (i < nw)                     w1b[i]            = (__bf16)w1[i];
  else if (i < 2 * nw)            w2b[i - nw]       = (__bf16)w2[i - nw];
  else if (i < 2 * nw + nkv)      kb[i - 2 * nw]    = (__bf16)k[i - 2 * nw];
  else if (i < 2 * nw + 2 * nkv)  vb[i - 2 * nw - nkv] = (__bf16)v[i - 2 * nw - nkv];
}

// ---------------------------------------------------------------- flash attention
// grid (N/64, B), 256 threads. O written bf16 token-major [b][n][c].
__global__ __launch_bounds__(256)
void attn_kernel(const float* __restrict__ qg, const __bf16* __restrict__ kb,
                 const __bf16* __restrict__ vb, __bf16* __restrict__ o_ws) {
  extern __shared__ char smem_a[];
  __bf16* Qs   = (__bf16*)smem_a;               // 64*QS
  __bf16* Ks   = Qs + 64 * QS;                  // 64*QS
  __bf16* Vs   = Ks + 64 * QS;                  // 512*VS  [c][m]
  float*  Sc   = (float*)(Vs + C_ * VS);        // 64*SS
  __bf16* Ps   = (__bf16*)(Sc + 64 * SS);       // 64*PSR
  float*  rmax = (float*)(Ps + 64 * PSR);
  float*  rsum = rmax + 64;
  float*  rscl = rsum + 64;

  const int b    = blockIdx.y;
  const int row0 = blockIdx.x * 64;
  const int tid  = threadIdx.x;
  const int wave = tid >> 5, lane = tid & 31;
  const int l15  = lane & 15, hi = lane >> 4;
  const float scale = 0.044194173824159216f;    // 512^-0.5

  // Q tile: global f32 [c][n] -> LDS bf16 [row][c] (transpose; TDM can't)
  for (int i = tid; i < 64 * C_; i += 256) {
    int c = i >> 6, r = i & 63;
    Qs[r * QS + c] = (__bf16)qg[((size_t)b * C_ + c) * N_ + row0 + r];
  }
  if (tid < 64) { rmax[tid] = -INFINITY; rsum[tid] = 0.f; }

  v8f acc[16] = {};                             // wave owns 64 channels: 4x4 tiles
  __syncthreads();

  for (int kbk = 0; kbk < N_ / 64; ++kbk) {
    const int k0g = kbk * 64;

    // V tile [c][m] via Tensor Data Mover: 512 rows x 64 bf16, global stride N_,
    // LDS padded to VS=80 (pad_interval: 32 DW -> code 4; pad_amount: 8 DW -> code 7)
    if (tid == 0) {
      tdm_load_2d((unsigned)(size_t)Vs,
                  vb + (size_t)b * C_ * N_ + k0g,
                  (1u << 16) | (1u << 20) | (4u << 22) | (7u << 25),
                  N_, C_, 64, C_, N_);
    }
    // K tile: bf16 [c][n] -> LDS [key][c] (transpose, manual)
    for (int i = tid; i < 64 * C_; i += 256) {
      int c = i >> 6, m = i & 63;
      Ks[m * QS + c] = kb[((size_t)b * C_ + c) * N_ + k0g + m];
    }
    if (kbk + 1 < N_ / 64) {                    // hint next K block toward cache
      __builtin_prefetch(&kb[((size_t)b * C_ + tid * 2) * N_ + k0g + 64], 0, 1);
    }
    __builtin_amdgcn_s_wait_tensorcnt(0);
    __syncthreads();

    // S = (Q . K^T) * scale : 64x64 = 16 tiles, 2 per wave, K-loop over 512
    for (int t = 0; t < 2; ++t) {
      int ti = wave * 2 + t;
      int tr = (ti >> 2) * 16, tc = (ti & 3) * 16;
      v8f s = {};
      for (int kc = 0; kc < C_ / 32; ++kc) {
        v16bf a  = frag_a(Qs, tr, QS, kc * 32, l15, hi);
        v16bf bb = frag_b(Ks, tc, QS, kc * 32, l15, hi);
        s = __builtin_amdgcn_wmma_f32_16x16x32_bf16(false, a, false, bb,
                                                    (short)0, s, false, false);
      }
      for (int r = 0; r < 8; ++r)
        Sc[(tr + r + hi * 8) * SS + tc + l15] = s[r] * scale;
    }
    __syncthreads();

    // online softmax: one thread per row
    if (tid < 64) {
      float mo = rmax[tid], mn = mo;
      for (int j = 0; j < 64; ++j) mn = fmaxf(mn, Sc[tid * SS + j]);
      float corr = __expf(mo - mn), sum = 0.f;
      for (int j = 0; j < 64; ++j) {
        float p = __expf(Sc[tid * SS + j] - mn);
        Ps[tid * PSR + j] = (__bf16)p;
        sum += p;
      }
      rsum[tid] = rsum[tid] * corr + sum;
      rmax[tid] = mn;
      rscl[tid] = corr;
    }
    __syncthreads();

    // rescale O, then O += P . V (wave owns channels wave*64 .. +63)
    for (int tr = 0; tr < 4; ++tr)
      for (int r = 0; r < 8; ++r) {
        float f = rscl[tr * 16 + r + hi * 8];
        for (int tc = 0; tc < 4; ++tc) acc[tr * 4 + tc][r] *= f;
      }
    for (int tr = 0; tr < 4; ++tr)
      for (int kc = 0; kc < 2; ++kc) {
        v16bf a = frag_a(Ps, tr * 16, PSR, kc * 32, l15, hi);
        for (int tc = 0; tc < 4; ++tc) {
          v16bf bb = frag_b(Vs, wave * 64 + tc * 16, VS, kc * 32, l15, hi);
          acc[tr * 4 + tc] = __builtin_amdgcn_wmma_f32_16x16x32_bf16(
              false, a, false, bb, (short)0, acc[tr * 4 + tc], false, false);
        }
      }
    __syncthreads();
  }

  // normalize + store bf16 [b][n][c] (lanes contiguous in c)
  for (int tr = 0; tr < 4; ++tr)
    for (int r = 0; r < 8; ++r) {
      int row = tr * 16 + r + hi * 8;
      float inv = 1.f / rsum[row];
      for (int tc = 0; tc < 4; ++tc) {
        int ch = wave * 64 + tc * 16 + l15;
        o_ws[((size_t)b * N_ + row0 + row) * C_ + ch] =
            (__bf16)(acc[tr * 4 + tc][r] * inv);
      }
    }
}

// ---------------------------------------------------------------- fused FFN
// grid (N/32, B), 256 threads. Hidden (2048 x 32) lives only in LDS.
__global__ __launch_bounds__(256)
void ffn_kernel(const __bf16* __restrict__ o_ws, const __bf16* __restrict__ w1b,
                const __bf16* __restrict__ w2b, const float* __restrict__ b1,
                const float* __restrict__ b2, const float* __restrict__ gamma,
                const float* __restrict__ query, float* __restrict__ out) {
  extern __shared__ char smem_f[];
  __bf16* Os  = (__bf16*)smem_f;               // 32*OSS  [t][c]
  __bf16* Hs  = Os + 32 * OSS;                 // 32*HSS  [t][d]
  float*  b1s = (float*)(Hs + 32 * HSS);       // 2048
  float*  b2s = b1s + C4_;                     // 512

  const int b   = blockIdx.y;
  const int n0  = blockIdx.x * 32;
  const int tid = threadIdx.x;
  const int wave = tid >> 5, lane = tid & 31;
  const int l15 = lane & 15, hi = lane >> 4;

  // Stage O tile + biases with the Tensor Data Mover (one op per wave 0/1/2).
  // O tile: 32 rows x 512 bf16, contiguous in global; LDS padded to OSS=528
  // (pad_interval: 256 DW -> code 7; pad_amount: 8 DW -> code 7).
  if (tid == 0) {
    tdm_load_2d((unsigned)(size_t)Os,
                o_ws + ((size_t)b * N_ + n0) * C_,
                (1u << 16) | (1u << 20) | (7u << 22) | (7u << 25),
                C_, (unsigned)(B_ * N_), C_, 32, C_);
  } else if (tid == 32) {
    tdm_load_2d((unsigned)(size_t)b1s, b1, (2u << 16), C4_, 1, C4_, 1, C4_);
  } else if (tid == 64) {
    tdm_load_2d((unsigned)(size_t)b2s, b2, (2u << 16), C_, 1, C_, 1, C_);
  }
  __builtin_amdgcn_s_wait_tensorcnt(0);
  __syncthreads();

  // GEMM1: hdn = relu(W1 . O + b1); D tile -> Hs[t][d] via packed b128 stores
  for (int rt = wave; rt < C4_ / 16; rt += 8) {
    int d0 = rt * 16;
    v8f h0 = {}, h1 = {};
    for (int kc = 0; kc < C_ / 32; ++kc) {
      v16bf a   = frag_a(w1b, d0, C_, kc * 32, l15, hi);
      v16bf bb0 = frag_b(Os, 0,  OSS, kc * 32, l15, hi);
      v16bf bb1 = frag_b(Os, 16, OSS, kc * 32, l15, hi);
      h0 = __builtin_amdgcn_wmma_f32_16x16x32_bf16(false, a, false, bb0,
                                                   (short)0, h0, false, false);
      h1 = __builtin_amdgcn_wmma_f32_16x16x32_bf16(false, a, false, bb1,
                                                   (short)0, h1, false, false);
    }
    Pack8 p0, p1;
    for (int r = 0; r < 8; ++r) {
      int d = d0 + hi * 8 + r;
      p0.h[r] = (__bf16)fmaxf(h0[r] + b1s[d], 0.f);
      p1.h[r] = (__bf16)fmaxf(h1[r] + b1s[d], 0.f);
    }
    *reinterpret_cast<uint4*>(&Hs[l15 * HSS + d0 + hi * 8])        = p0.u;
    *reinterpret_cast<uint4*>(&Hs[(16 + l15) * HSS + d0 + hi * 8]) = p1.u;
  }
  __syncthreads();

  // GEMM2: ffn = W2 . hdn; out = (ffn + b2) * gamma + query
  const float g = gamma[0];
  for (int rt = wave * 4; rt < wave * 4 + 4; ++rt) {
    int c0 = rt * 16;
    v8f f0 = {}, f1 = {};
    for (int kc = 0; kc < C4_ / 32; ++kc) {
      v16bf a   = frag_a(w2b, c0, C4_, kc * 32, l15, hi);
      v16bf bb0 = frag_b(Hs, 0,  HSS, kc * 32, l15, hi);
      v16bf bb1 = frag_b(Hs, 16, HSS, kc * 32, l15, hi);
      f0 = __builtin_amdgcn_wmma_f32_16x16x32_bf16(false, a, false, bb0,
                                                   (short)0, f0, false, false);
      f1 = __builtin_amdgcn_wmma_f32_16x16x32_bf16(false, a, false, bb1,
                                                   (short)0, f1, false, false);
    }
    for (int r = 0; r < 8; ++r) {
      int c = c0 + r + hi * 8;
      size_t base = ((size_t)b * C_ + c) * N_ + n0;
      out[base + l15]      = (f0[r] + b2s[c]) * g + query[base + l15];
      out[base + 16 + l15] = (f1[r] + b2s[c]) * g + query[base + 16 + l15];
    }
  }
}

extern "C" void kernel_launch(void* const* d_in, const int* in_sizes, int n_in,
                              void* d_out, int out_size, void* d_ws, size_t ws_size,
                              hipStream_t stream) {
  const float* q  = (const float*)d_in[0];
  const float* k  = (const float*)d_in[1];
  const float* v  = (const float*)d_in[2];
  const float* w1 = (const float*)d_in[3];
  const float* b1 = (const float*)d_in[4];
  const float* w2 = (const float*)d_in[5];
  const float* b2 = (const float*)d_in[6];
  const float* gm = (const float*)d_in[7];
  float* out = (float*)d_out;

  // workspace: W1bf(2MB) | W2bf(2MB) | O bf16(32MB) | K bf16(32MB) | V bf16(32MB)
  const size_t nw  = (size_t)C4_ * C_;
  const size_t nkv = (size_t)B_ * C_ * N_;
  __bf16* w1b  = (__bf16*)d_ws;
  __bf16* w2b  = w1b + nw;
  __bf16* o_ws = w2b + nw;
  __bf16* kbuf = o_ws + nkv;
  __bf16* vbuf = kbuf + nkv;

  const size_t ncvt = 2 * nw + 2 * nkv;
  cvt_kernel<<<(unsigned)((ncvt + 255) / 256), 256, 0, stream>>>(
      w1, w2, k, v, w1b, w2b, kbuf, vbuf);

  const size_t attn_smem =
      (size_t)(64 * QS + 64 * QS + C_ * VS) * 2 +  // Qs, Ks, Vs
      (size_t)64 * SS * 4 +                        // Sc
      (size_t)64 * PSR * 2 +                       // Ps
      3 * 64 * 4;                                  // rmax/rsum/rscl
  attn_kernel<<<dim3(N_ / 64, B_), 256, attn_smem, stream>>>(q, kbuf, vbuf, o_ws);

  const size_t ffn_smem =
      (size_t)(32 * OSS + 32 * HSS) * 2 + (size_t)(C4_ + C_) * 4;
  ffn_kernel<<<dim3(N_ / 32, B_), 256, ffn_smem, stream>>>(o_ws, w1b, w2b,
                                                           b1, b2, gm, q, out);
}
